// GraphAttentionLayer_31233002177083
// MI455X (gfx1250) — matrix operand order repaired and checked by reading, not compiled, not run on hardware
//
#include <hip/hip_runtime.h>

// ---------------------------------------------------------------------------
// Graph attention on gfx1250 (MI455X), all matmuls via v_wmma_f32_16x16x32_bf16
// K/V tile staging via CDNA5 async global->LDS copies (ASYNCcnt path).
// B=4, N=4096, H=4, D=32, model dim 128.
// ---------------------------------------------------------------------------

#define B_DIM 4
#define N_DIM 4096
#define H_DIM 4
#define D_DIM 32
#define C_DIM 128          // H*D
#define BN (B_DIM * N_DIM) // 16384
#define KC 64              // keys per chunk in attention kernel
#define LDS_STRIDE 72      // 64 + 8 pad -> 144B rows, conflict-free b128 reads

typedef __attribute__((ext_vector_type(16))) __bf16 v16bf;
typedef __attribute__((ext_vector_type(8)))  float  v8f;

union U16 { v16bf v; uint4 u[2]; };

// 1/sqrt(32) * log2(e): softmax done in exp2 domain
#define K_SCALE 0.25505402f

__device__ inline float rowmax16(float v) {
#pragma unroll
  for (int m = 1; m < 16; m <<= 1) v = fmaxf(v, __shfl_xor(v, m, 32));
  return v;
}
__device__ inline float rowsum16(float v) {
#pragma unroll
  for (int m = 1; m < 16; m <<= 1) v += __shfl_xor(v, m, 32);
  return v;
}

// CDNA5 async global->LDS copy, 16B per lane. LDS destination is the low 32
// bits of the generic pointer (LDS aperture addressing, ISA 10.2). Tracked by
// ASYNCcnt; pair with s_wait_asynccnt before the workgroup barrier.
__device__ inline void async_copy16(__bf16* lds_dst, const __bf16* g_src) {
  const unsigned lds_off = (unsigned)(uintptr_t)lds_dst;
  asm volatile("global_load_async_to_lds_b128 %0, %1, off"
               :
               : "v"(lds_off), "v"(g_src)
               : "memory");
}
__device__ inline void wait_async0() {
  asm volatile("s_wait_asynccnt 0" ::: "memory");
}

// ---------------------------------------------------------------------------
// Kernel 1: Q/K/V projections. One wave per 16x16 output tile.
// Q,K stored (B,H,N,D) bf16 row-major; V stored transposed (B,H,D,N) bf16 so
// the attention kernel's B-fragments are contiguous 16B loads.
// ---------------------------------------------------------------------------
__global__ __launch_bounds__(32) void qkv_kernel(
    const float* __restrict__ x,
    const float* __restrict__ Wq, const float* __restrict__ Wk,
    const float* __restrict__ Wv,
    __bf16* __restrict__ q, __bf16* __restrict__ k, __bf16* __restrict__ vt) {
  const int lane = threadIdx.x & 31;
  const int col  = lane & 15;
  const int hb   = lane >> 4;
  const int rowbase = blockIdx.x * 16;  // row in flattened (B*N)
  const int cbase   = blockIdx.y * 16;  // output column tile
  const int mat     = blockIdx.z;       // 0=Q 1=K 2=V
  const float* W = (mat == 0) ? Wq : ((mat == 1) ? Wk : Wv);

  const float* xrow = x + (size_t)(rowbase + col) * C_DIM;
  v8f acc = {};
#pragma unroll
  for (int kk = 0; kk < 4; ++kk) {
    // A fragment (16x32 bf16): lane holds row=col, K runs [base,base+16)
    // with base = hb*16 (contiguous, matches B).
    U16 Af;
#pragma unroll
    for (int i = 0; i < 8; ++i)
      Af.v[i] = (__bf16)xrow[kk * 32 + hb * 16 + i];
#pragma unroll
    for (int i = 0; i < 8; ++i)
      Af.v[8 + i] = (__bf16)xrow[kk * 32 + hb * 16 + 8 + i];
    // B fragment (32x16 bf16): lane holds col, K = hb*16 + i.
    U16 Bf;
#pragma unroll
    for (int i = 0; i < 16; ++i)
      Bf.v[i] = (__bf16)W[(size_t)(kk * 32 + hb * 16 + i) * C_DIM + cbase + col];
    acc = __builtin_amdgcn_wmma_f32_16x16x32_bf16(
        false, Af.v, false, Bf.v, (short)0, acc, false, false);
  }

  const int c = cbase + col;
  const int h = c >> 5, d = c & 31;
#pragma unroll
  for (int r = 0; r < 8; ++r) {
    const int bn = rowbase + r + hb * 8;   // D-layout: row = r + 8*(lane/16)
    const int b  = bn >> 12, n = bn & (N_DIM - 1);
    const __bf16 val = (__bf16)acc[r];
    if (mat == 2) {
      vt[(((size_t)(b * H_DIM + h)) * D_DIM + d) * N_DIM + n] = val;
    } else {
      __bf16* dst = (mat == 0) ? q : k;
      dst[(((size_t)(b * H_DIM + h)) * N_DIM + n) * D_DIM + d] = val;
    }
  }
}

// ---------------------------------------------------------------------------
// Kernel 2: flash attention with adjacency mask.
// Block = 256 threads (8 waves); block owns 128 query rows of one (b,h);
// wave owns 16 query rows. Loop keys in chunks of 64.
// ---------------------------------------------------------------------------
__global__ __launch_bounds__(256) void attn_kernel(
    const __bf16* __restrict__ q, const __bf16* __restrict__ k,
    const __bf16* __restrict__ vt, const int* __restrict__ adj,
    __bf16* __restrict__ o_pre) {
  __shared__ __bf16 k_s[KC * LDS_STRIDE];        // 64 keys x 32 d (padded)
  __shared__ __bf16 v_s[D_DIM * LDS_STRIDE];     // V^T: 32 d x 64 keys (padded)
  __shared__ __bf16 p_s[8 * 16 * LDS_STRIDE];    // per-wave P: 16 x 64 (padded)

  const int bh = blockIdx.x;
  const int b  = bh >> 2, h = bh & 3;
  const int nbase = blockIdx.y * 128;
  const int tid  = threadIdx.x;
  const int lane = tid & 31, wave = tid >> 5;
  const int col  = lane & 15, hb = lane >> 4;

  const __bf16* qb = q  + (size_t)bh * N_DIM * D_DIM;
  const __bf16* kb = k  + (size_t)bh * N_DIM * D_DIM;
  const __bf16* vb = vt + (size_t)bh * N_DIM * D_DIM;

  // Persistent Q fragment for this wave's 16 rows.
  const int qn = nbase + wave * 16 + col;
  const __bf16* qrow = qb + (size_t)qn * D_DIM + hb * 16;
  U16 Qf;
  Qf.u[0] = *(const uint4*)qrow;
  Qf.u[1] = *(const uint4*)(qrow + 8);

  // Per-thread staging coordinates (constant across chunks).
  const int krow = tid >> 2, piece = tid & 3;  // K: 64 rows x 4 x 16B
  const int drow = tid >> 3, p2 = tid & 7;     // V^T: 32 rows x 8 x 16B
  __bf16* k_dst = k_s + krow * LDS_STRIDE + piece * 8;
  __bf16* v_dst = v_s + drow * LDS_STRIDE + p2 * 8;

  float mrow[8], lrow[8];
  v8f o0 = {}, o1 = {};   // O accumulator, cols d 0..15 / 16..31
#pragma unroll
  for (int r = 0; r < 8; ++r) { mrow[r] = -1e30f; lrow[r] = 0.f; }

  for (int kc = 0; kc < N_DIM; kc += KC) {
    // --- async staging: K chunk (4KB contiguous) and V^T chunk -> LDS ---
    async_copy16(k_dst, kb + (size_t)(kc + krow) * D_DIM + piece * 8);
    async_copy16(v_dst, vb + (size_t)drow * N_DIM + kc + p2 * 8);
    wait_async0();        // our own async writes have landed in LDS
    __syncthreads();      // everyone's writes are visible

    // --- scores: S = Q (16x32) x K^T, four 16-key tiles ---
    v8f s[4];
#pragma unroll
    for (int kt = 0; kt < 4; ++kt) {
      U16 Bf;  // B[d][key]: lane=key, 16 contiguous d at hb*16
      const __bf16* kr = k_s + (kt * 16 + col) * LDS_STRIDE + hb * 16;
      Bf.u[0] = *(const uint4*)kr;
      Bf.u[1] = *(const uint4*)(kr + 8);
      v8f z = {};
      s[kt] = __builtin_amdgcn_wmma_f32_16x16x32_bf16(
          false, Qf.v, false, Bf.v, (short)0, z, false, false);
    }

    // --- adjacency mask + online softmax (exp2 domain) ---
#pragma unroll
    for (int r = 0; r < 8; ++r) {
      const int n = nbase + wave * 16 + r + hb * 8;
      const int* arow = adj + (size_t)n * N_DIM + kc + col;
      float sv[4];
#pragma unroll
      for (int kt = 0; kt < 4; ++kt) {
        const int a = arow[kt * 16];
        const float x = s[kt][r];
        sv[kt] = a ? x * K_SCALE : -1e30f;
      }
      float rm = fmaxf(fmaxf(sv[0], sv[1]), fmaxf(sv[2], sv[3]));
      rm = rowmax16(rm);
      const float mo = mrow[r];
      const float mn = fmaxf(mo, rm);
      const float corr = exp2f(mo - mn);
      mrow[r] = mn;
      float rs = 0.f;
#pragma unroll
      for (int kt = 0; kt < 4; ++kt) {
        float p = (sv[kt] <= -1e29f) ? 0.f : exp2f(sv[kt] - mn);
        s[kt][r] = p;
        rs += p;
      }
      rs = rowsum16(rs);
      lrow[r] = lrow[r] * corr + rs;
      o0[r] *= corr;
      o1[r] *= corr;
      // bounce P through LDS (D-layout -> A-fragment layout)
#pragma unroll
      for (int kt = 0; kt < 4; ++kt)
        p_s[(wave * 16 + r + hb * 8) * LDS_STRIDE + kt * 16 + col] =
            (__bf16)s[kt][r];
    }
    asm volatile("s_wait_dscnt 0" ::: "memory");  // same-wave LDS RAW

    // --- O += P (16x64) x V (64x32), two K=32 steps x two 16-col halves ---
#pragma unroll
    for (int kk = 0; kk < 2; ++kk) {
      U16 Af;
      const __bf16* pr =
          p_s + (wave * 16 + col) * LDS_STRIDE + kk * 32 + hb * 16;
      Af.u[0] = *(const uint4*)pr;
      Af.u[1] = *(const uint4*)(pr + 8);
      U16 B0, B1;  // from V^T rows: contiguous 16 keys per lane
      const __bf16* v0 = v_s + col * LDS_STRIDE + kk * 32 + hb * 16;
      B0.u[0] = *(const uint4*)v0;
      B0.u[1] = *(const uint4*)(v0 + 8);
      const __bf16* v1 = v_s + (16 + col) * LDS_STRIDE + kk * 32 + hb * 16;
      B1.u[0] = *(const uint4*)v1;
      B1.u[1] = *(const uint4*)(v1 + 8);
      o0 = __builtin_amdgcn_wmma_f32_16x16x32_bf16(
          false, Af.v, false, B0.v, (short)0, o0, false, false);
      o1 = __builtin_amdgcn_wmma_f32_16x16x32_bf16(
          false, Af.v, false, B1.v, (short)0, o1, false, false);
    }
    __syncthreads();
  }

  // --- normalize and store pre-projection output (B, N, 128) bf16 ---
#pragma unroll
  for (int r = 0; r < 8; ++r) {
    const float inv = 1.0f / lrow[r];
    const int n = nbase + wave * 16 + r + hb * 8;
    const size_t base = ((size_t)b * N_DIM + n) * C_DIM + h * D_DIM;
    o_pre[base + col]      = (__bf16)(o0[r] * inv);
    o_pre[base + 16 + col] = (__bf16)(o1[r] * inv);
  }
}

// ---------------------------------------------------------------------------
// Kernel 3: output projection  out = o_pre @ W_o + b_o  (f32 store)
// ---------------------------------------------------------------------------
__global__ __launch_bounds__(32) void oproj_kernel(
    const __bf16* __restrict__ o_pre, const float* __restrict__ Wo,
    const float* __restrict__ bo, float* __restrict__ out) {
  const int lane = threadIdx.x & 31;
  const int col  = lane & 15;
  const int hb   = lane >> 4;
  const int rowbase = blockIdx.x * 16;
  const int cbase   = blockIdx.y * 16;

  const __bf16* arow = o_pre + (size_t)(rowbase + col) * C_DIM;
  v8f acc = {};
#pragma unroll
  for (int kk = 0; kk < 4; ++kk) {
    U16 Af;
    const __bf16* p0 = arow + kk * 32 + hb * 16;
    Af.u[0] = *(const uint4*)p0;
    Af.u[1] = *(const uint4*)(p0 + 8);
    U16 Bf;
#pragma unroll
    for (int i = 0; i < 16; ++i)
      Bf.v[i] = (__bf16)Wo[(size_t)(kk * 32 + hb * 16 + i) * C_DIM + cbase + col];
    acc = __builtin_amdgcn_wmma_f32_16x16x32_bf16(
        false, Af.v, false, Bf.v, (short)0, acc, false, false);
  }
  const float bias = bo[cbase + col];
#pragma unroll
  for (int r = 0; r < 8; ++r)
    out[(size_t)(rowbase + r + hb * 8) * C_DIM + cbase + col] = acc[r] + bias;
}

// ---------------------------------------------------------------------------
extern "C" void kernel_launch(void* const* d_in, const int* in_sizes, int n_in,
                              void* d_out, int out_size, void* d_ws,
                              size_t ws_size, hipStream_t stream) {
  const float* x   = (const float*)d_in[0];
  const int*   adj = (const int*)d_in[1];
  const float* Wq  = (const float*)d_in[2];
  const float* Wk  = (const float*)d_in[3];
  const float* Wv  = (const float*)d_in[4];
  const float* Wo  = (const float*)d_in[5];
  const float* bo  = (const float*)d_in[6];
  float* out = (float*)d_out;

  // Workspace: 4 bf16 buffers of B*N*128 = 2,097,152 elements (4 MiB each).
  const size_t elems = (size_t)BN * C_DIM;
  __bf16* qw = (__bf16*)d_ws;
  __bf16* kw = qw + elems;
  __bf16* vw = kw + elems;
  __bf16* ow = vw + elems;

  qkv_kernel<<<dim3(BN / 16, C_DIM / 16, 3), 32, 0, stream>>>(
      x, Wq, Wk, Wv, qw, kw, vw);
  attn_kernel<<<dim3(B_DIM * H_DIM, N_DIM / 128), 256, 0, stream>>>(
      qw, kw, vw, adj, ow);
  oproj_kernel<<<dim3(BN / 16, C_DIM / 16), 32, 0, stream>>>(ow, Wo, bo, out);
}
